// MoELinear_62311385530886
// MI455X (gfx1250) — matrix operand order repaired and checked by reading, not compiled, not run on hardware
//
#include <hip/hip_runtime.h>
#include <hip/hip_bf16.h>

// ---------------------------------------------------------------------------
// MoE top-2 linear for MI455X (gfx1250, wave32).
//   out[t,:] = sum_{k in top2(t)} softmax_w[k] * (x[t,:] @ We[e_k].T)
// Gate+route (compaction) -> expert-grouped bf16 WMMA GEMM with gather A /
// scatter-atomic epilogue.  Block tile 128x128, wave tile 32x64 (8 WMMA/iter).
// ---------------------------------------------------------------------------

#define N_TOK   8192      // B*T
#define CIN     1024
#define COUT    4096
#define NEXP    8
#define CAP     8192      // max tokens per expert (top-2 indices distinct)

#define TILE_M  128
#define TILE_N  128
#define KSTEP   32        // matches WMMA K for bf16
#define LDT     40        // LDS row stride in bf16 elems (80 B, 16B-aligned chunks)

typedef __attribute__((ext_vector_type(16))) __bf16 v16bf;
typedef __attribute__((ext_vector_type(8)))  float  v8f;

union Frag {              // one WMMA A/B operand: 8 VGPRs per lane
    v16bf v;
    uint4 q[2];
};

union Pack16 {            // 16 bf16 staged for two ds_store_b128
    __bf16 h[16];
    uint4  q[2];
};

// ---------------------------------------------------------------------------
// Kernel 1: gating.  One wave per token; 8 logits via lane-strided dots,
// butterfly reduce, top-2 + 2-way softmax, atomic-append to expert lists.
// ---------------------------------------------------------------------------
__global__ __launch_bounds__(256) void moe_gate_kernel(
    const float* __restrict__ x, const float* __restrict__ Wg,
    const float* __restrict__ bg, int* __restrict__ cnt,
    int* __restrict__ tok_list, float* __restrict__ w_list)
{
    const int wave = threadIdx.x >> 5;
    const int lane = threadIdx.x & 31;
    const int t    = blockIdx.x * 8 + wave;
    if (t >= N_TOK) return;

    const float* xr = x + (size_t)t * CIN;

    float acc[NEXP];
#pragma unroll
    for (int e = 0; e < NEXP; ++e) acc[e] = 0.0f;

    for (int k = lane; k < CIN; k += 32) {
        const float xv = xr[k];
#pragma unroll
        for (int e = 0; e < NEXP; ++e)
            acc[e] = fmaf(xv, Wg[e * CIN + k], acc[e]);
    }

#pragma unroll
    for (int e = 0; e < NEXP; ++e) {
#pragma unroll
        for (int off = 16; off > 0; off >>= 1)
            acc[e] += __shfl_xor(acc[e], off, 32);
        acc[e] += bg[e];
    }

    if (lane == 0) {
        int   i0 = 0; float m0 = acc[0];
#pragma unroll
        for (int e = 1; e < NEXP; ++e)
            if (acc[e] > m0) { m0 = acc[e]; i0 = e; }
        int   i1 = -1; float m1 = -3.4e38f;
#pragma unroll
        for (int e = 0; e < NEXP; ++e)
            if (e != i0 && acc[e] > m1) { m1 = acc[e]; i1 = e; }

        const float e1 = __expf(m1 - m0);
        const float s  = 1.0f / (1.0f + e1);

        int p0 = atomicAdd(&cnt[i0], 1);
        tok_list[i0 * CAP + p0] = t;  w_list[i0 * CAP + p0] = s;
        int p1 = atomicAdd(&cnt[i1], 1);
        tok_list[i1 * CAP + p1] = t;  w_list[i1 * CAP + p1] = e1 * s;
    }
}

// ---------------------------------------------------------------------------
// Kernel 2: expert-grouped GEMM.  256 threads / 8 waves, tile 128x128.
// Waves 4(M) x 2(N); each wave owns 32x64 = eight 16x16 f32 accumulators.
// ---------------------------------------------------------------------------
__global__ __launch_bounds__(256) void moe_gemm_kernel(
    const float* __restrict__ x, const float* __restrict__ We,
    const int* __restrict__ cnt, const int* __restrict__ tok_list,
    const float* __restrict__ w_list, float* __restrict__ out)
{
    const int e  = blockIdx.z;
    const int mt = blockIdx.y;
    const int nb = blockIdx.x * TILE_N;

    const int count = cnt[e];
    if (mt * TILE_M >= count) return;   // uniform early exit

    __shared__ __align__(16) __bf16 As[TILE_M * LDT];
    __shared__ __align__(16) __bf16 Bs[TILE_N * LDT];
    __shared__ int   sTok[TILE_M];
    __shared__ float sW[TILE_M];

    const int tid = threadIdx.x;

    if (tid < TILE_M) {
        const int slot = mt * TILE_M + tid;
        const bool v   = slot < count;
        sTok[tid] = v ? tok_list[e * CAP + slot] : 0;
        sW[tid]   = v ? w_list[e * CAP + slot]   : 0.0f;
    }
    __syncthreads();

    const int lane = tid & 31;
    const int wave = tid >> 5;
    const int wm   = wave & 3;   // 4 waves across M -> 32 rows each
    const int wn   = wave >> 2;  // 2 waves across N -> 64 cols each
    const int half = lane >> 4;  // ISA lane-group (K split)
    const int l16  = lane & 15;

    // staging: 128 rows x 32 K f32 per operand; thread = (row, 16-float half)
    const int arow = tid >> 1, apart = tid & 1;

    const float* aptr = x + (size_t)sTok[arow] * CIN + apart * 16;
    const float* bptr = We + (size_t)e * COUT * CIN
                           + (size_t)(nb + arow) * CIN + apart * 16;

    v8f c[2][4];
#pragma unroll
    for (int mf = 0; mf < 2; ++mf)
#pragma unroll
        for (int nf = 0; nf < 4; ++nf) c[mf][nf] = (v8f)(0.0f);

    for (int k0 = 0; k0 < CIN; k0 += KSTEP) {
        // ---- global f32 loads (b128) + prefetch next K chunk ----
        const float4 a0 = *(const float4*)(aptr + k0 + 0);
        const float4 a1 = *(const float4*)(aptr + k0 + 4);
        const float4 a2 = *(const float4*)(aptr + k0 + 8);
        const float4 a3 = *(const float4*)(aptr + k0 + 12);
        const float4 b0 = *(const float4*)(bptr + k0 + 0);
        const float4 b1 = *(const float4*)(bptr + k0 + 4);
        const float4 b2 = *(const float4*)(bptr + k0 + 8);
        const float4 b3 = *(const float4*)(bptr + k0 + 12);
        if (k0 + KSTEP < CIN) {
            __builtin_prefetch(aptr + k0 + KSTEP, 0, 1);  // global_prefetch_b8
            __builtin_prefetch(bptr + k0 + KSTEP, 0, 1);
        }

        __syncthreads();   // previous tile fully consumed

        // ---- native f32->bf16 converts (v_cvt_*bf16*), store to LDS ----
        {
            Pack16 pa, pb;
            const float av[16] = {a0.x,a0.y,a0.z,a0.w, a1.x,a1.y,a1.z,a1.w,
                                  a2.x,a2.y,a2.z,a2.w, a3.x,a3.y,a3.z,a3.w};
            const float bv[16] = {b0.x,b0.y,b0.z,b0.w, b1.x,b1.y,b1.z,b1.w,
                                  b2.x,b2.y,b2.z,b2.w, b3.x,b3.y,b3.z,b3.w};
#pragma unroll
            for (int j = 0; j < 16; ++j) pa.h[j] = (__bf16)av[j];
#pragma unroll
            for (int j = 0; j < 16; ++j) pb.h[j] = (__bf16)bv[j];

            uint4* da = (uint4*)&As[arow * LDT + apart * 16];
            uint4* db = (uint4*)&Bs[arow * LDT + apart * 16];
            da[0] = pa.q[0];  da[1] = pa.q[1];
            db[0] = pb.q[0];  db[1] = pb.q[1];
        }
        __syncthreads();   // tile ready

        // ---- fragments per ISA 7.12.2, eight WMMAs ----
        Frag a[2], b[4];
#pragma unroll
        for (int mf = 0; mf < 2; ++mf) {
            const int row = wm * 32 + mf * 16 + l16;
            // A 16x32: lanes 0-15 K 0..7 & 16..23; lanes 16-31 K 8..15 & 24..31
            a[mf].q[0] = *(const uint4*)&As[row * LDT + half * 8];
            a[mf].q[1] = *(const uint4*)&As[row * LDT + 16 + half * 8];
        }
#pragma unroll
        for (int nf = 0; nf < 4; ++nf) {
            const int col = wn * 64 + nf * 16 + l16;
            // B 32x16: lanes 0-15 K 0..7 & 16..23; lanes 16-31 K 8..15 & 24..31 (col = We row)
            b[nf].q[0] = *(const uint4*)&Bs[col * LDT + half * 8];
            b[nf].q[1] = *(const uint4*)&Bs[col * LDT + 16 + half * 8];
        }
#pragma unroll
        for (int mf = 0; mf < 2; ++mf)
#pragma unroll
            for (int nf = 0; nf < 4; ++nf)
                c[mf][nf] = __builtin_amdgcn_wmma_f32_16x16x32_bf16(
                    false, a[mf].v, false, b[nf].v,
                    (short)0, c[mf][nf], false, false);
    }

    // ---- epilogue: scale by routing weight, scatter-accumulate ----
    const int rem = count - mt * TILE_M;
#pragma unroll
    for (int mf = 0; mf < 2; ++mf) {
#pragma unroll
        for (int r = 0; r < 8; ++r) {
            const int lr = wm * 32 + mf * 16 + half * 8 + r;  // D: VGPR r -> M
            if (lr < rem) {
                const int   tokn = sTok[lr];
                const float w    = sW[lr];
                float* orow = out + (size_t)tokn * COUT + nb + wn * 64 + l16;
#pragma unroll
                for (int nf = 0; nf < 4; ++nf)
                    atomicAdd(orow + nf * 16, w * c[mf][nf][r]);
            }
        }
    }
}

// ---------------------------------------------------------------------------
// Host launcher
// ---------------------------------------------------------------------------
extern "C" void kernel_launch(void* const* d_in, const int* in_sizes, int n_in,
                              void* d_out, int out_size, void* d_ws, size_t ws_size,
                              hipStream_t stream) {
    const float* x  = (const float*)d_in[0];   // [4,2048,1024]
    const float* We = (const float*)d_in[1];   // [8,4096,1024]
    const float* Wg = (const float*)d_in[2];   // [8,1024]
    const float* bg = (const float*)d_in[3];   // [8]
    float* out = (float*)d_out;                // [8192,4096]

    char*  ws       = (char*)d_ws;
    int*   cnt      = (int*)ws;                          // 8 ints (256 B pad)
    int*   tok_list = (int*)(ws + 256);                  // 8*8192 ints
    float* w_list   = (float*)(ws + 256 + NEXP * CAP * sizeof(int));

    hipMemsetAsync(d_out, 0, (size_t)out_size * sizeof(float), stream);
    hipMemsetAsync(cnt, 0, NEXP * sizeof(int), stream);

    moe_gate_kernel<<<N_TOK / 8, 256, 0, stream>>>(x, Wg, bg, cnt, tok_list,
                                                   w_list);

    dim3 grid(COUT / TILE_N, CAP / TILE_M, NEXP);  // inactive (e,mt) exit fast
    moe_gemm_kernel<<<grid, 256, 0, stream>>>(x, We, cnt, tok_list, w_list, out);
}
